// QGM_Decoder_51213190037773
// MI455X (gfx1250) — matrix operand order repaired and checked by reading, not compiled, not run on hardware
//
#include <hip/hip_runtime.h>
#include <hip/hip_bf16.h>
#include <math.h>

// Sizes fixed by the reference
#define BB 32
#define SS 2048
#define EE 512
#define NCC 128
#define NTT 32
#define HH 8
#define QQ 4

typedef __attribute__((ext_vector_type(16))) _Float16 v16h;
typedef __attribute__((ext_vector_type(8)))  float    v8f;

// ---------------------------------------------------------------------------
// K0: gather + mean embeddings -> head_emb, body_emb, boxop_e (each [B,E])
// ---------------------------------------------------------------------------
__global__ void embed_kernel(const float* __restrict__ enc_col,
                             const float* __restrict__ enc_tab,
                             const int* __restrict__ head_agg,
                             const int* __restrict__ head_col,
                             const int* __restrict__ quant_tab,
                             const int* __restrict__ box_op,
                             const float* __restrict__ agg_emb,
                             const float* __restrict__ box_op_emb,
                             float* __restrict__ head_out,
                             float* __restrict__ body_out,
                             float* __restrict__ boxop_out) {
    int b = blockIdx.x;
    int e = threadIdx.x;                       // 512 threads
    float hacc = 0.f;
#pragma unroll
    for (int h = 0; h < HH; ++h) {
        int a = head_agg[b * HH + h];
        int c = head_col[b * HH + h];
        hacc += agg_emb[a * EE + e] + enc_col[((size_t)b * NCC + c) * EE + e];
    }
    head_out[b * EE + e] = hacc * (1.f / (float)HH);
    float bacc = 0.f;
#pragma unroll
    for (int q = 0; q < QQ; ++q) {
        int t = quant_tab[b * QQ + q];
        bacc += enc_tab[((size_t)b * NTT + t) * EE + e];
    }
    body_out[b * EE + e] = bacc * (1.f / (float)QQ);
    boxop_out[b * EE + e] = box_op_emb[box_op[b] * EE + e];
}

// ---------------------------------------------------------------------------
// Generic WMMA GEMM:  out[M=32, N] = relu?( A[32,Ktot] * W^T + bias )
// A is given as up to three 512-column segments (row stride 512 each);
// segment index = k >> 9 (all K chunks are 32-aligned, segments 512-aligned).
// W row-major [N, Kw]; switches from (B0,ldb0) to (B1,ldb1) at k == kSplit.
// One wave computes one 16x16 D tile via v_wmma_f32_16x16x32_f16.
// ---------------------------------------------------------------------------
__global__ void wmma_gemm_kernel(const float* __restrict__ A0,
                                 const float* __restrict__ A1,
                                 const float* __restrict__ A2,
                                 const float* __restrict__ B0, int ldb0,
                                 int kSplit,
                                 const float* __restrict__ B1, int ldb1,
                                 const float* __restrict__ bias0,
                                 const float* __restrict__ bias1,
                                 int Ktot, int N,
                                 float* __restrict__ out, int applyRelu) {
    const int lane = threadIdx.x & 31;
    const int wave = threadIdx.x >> 5;
    const int tile = blockIdx.x * (blockDim.x >> 5) + wave;
    const int nTiles = N >> 4;
    const int mTile = tile / nTiles;           // 0 or 1 (M = 32)
    const int nTile = tile % nTiles;
    if (mTile >= 2) return;                    // wave-uniform; EXEC stays all-ones

    const int mBase = mTile << 4;
    const int nBase = nTile << 4;
    const int m = mBase + (lane & 15);
    const int n = nBase + (lane & 15);

    v8f acc = {0.f, 0.f, 0.f, 0.f, 0.f, 0.f, 0.f, 0.f};

    for (int kBase = 0; kBase < Ktot; kBase += 32) {
        // ---- A fragment (16x32 f16, 16 contiguous k per lane half) ----
        const int seg = kBase >> 9;
        const float* As = (seg == 0) ? A0 : (seg == 1) ? A1 : A2;
        const int kk = kBase & 511;
        const float* pa = As + (size_t)m * EE + kk + ((lane >> 4) << 4);
        v16h a;
#pragma unroll
        for (int j = 0; j < 8; ++j) a[j] = (_Float16)pa[j];
#pragma unroll
        for (int j = 0; j < 8; ++j) a[8 + j] = (_Float16)pa[8 + j];

        // ---- B fragment (32x16): B[k][n] = W[n*ldb + k], 16 contiguous k ----
        const float* Bw; int ldb, kb;
        if (kBase < kSplit) { Bw = B0; ldb = ldb0; kb = kBase; }
        else               { Bw = B1; ldb = ldb1; kb = kBase - kSplit; }
        const float* pb = Bw + (size_t)n * ldb + kb + ((lane >> 4) << 4);
        v16h bf;
#pragma unroll
        for (int j = 0; j < 16; ++j) bf[j] = (_Float16)pb[j];

        acc = __builtin_amdgcn_wmma_f32_16x16x32_f16(
            false, a, false, bf, (short)0, acc, false, false);
    }

    float bv = 0.f;
    if (bias0) bv += bias0[n];
    if (bias1) bv += bias1[n];
#pragma unroll
    for (int r = 0; r < 8; ++r) {
        int mm = mBase + r + ((lane >> 4) << 3);
        float v = acc[r] + bv;
        if (applyRelu) v = fmaxf(v, 0.f);
        out[(size_t)mm * N + n] = v;
    }
}

// ---------------------------------------------------------------------------
// K3: LSTM cell elementwise (torch gate order i,f,g,o), biases added here
// ---------------------------------------------------------------------------
__device__ __forceinline__ float sigf(float x) { return 1.f / (1.f + __expf(-x)); }

__global__ void lstm_cell_kernel(const float* __restrict__ gates,
                                 const float* __restrict__ b_ih,
                                 const float* __restrict__ b_hh,
                                 const float* __restrict__ c0,
                                 float* __restrict__ hout) {
    int idx = blockIdx.x * blockDim.x + threadIdx.x;   // b*E + e
    int b = idx >> 9, e = idx & 511;
    const float* g = gates + (size_t)b * (4 * EE);
    float gi = g[e]            + b_ih[e]            + b_hh[e];
    float gf = g[EE + e]       + b_ih[EE + e]       + b_hh[EE + e];
    float gg = g[2 * EE + e]   + b_ih[2 * EE + e]   + b_hh[2 * EE + e];
    float go = g[3 * EE + e]   + b_ih[3 * EE + e]   + b_hh[3 * EE + e];
    float c = sigf(gf) * c0[idx] + sigf(gi) * tanhf(gg);
    hout[idx] = sigf(go) * tanhf(c);
}

// ---------------------------------------------------------------------------
// K4: scores[b,s] = dot(src_att[b,s,:], h[b,:]) ; masked -> -inf
// one wave per s, lanes stride over E with float4 (coalesced 512B/wave/iter)
// ---------------------------------------------------------------------------
__global__ void scores_kernel(const float* __restrict__ src_att,
                              const float* __restrict__ h,
                              const unsigned char* __restrict__ mask,
                              float* __restrict__ scores) {
    int lane = threadIdx.x & 31;
    int wave = threadIdx.x >> 5;
    int b = blockIdx.y;
    int s = blockIdx.x * 8 + wave;
    const float* row = src_att + ((size_t)b * SS + s) * EE;
    const float* hv  = h + (size_t)b * EE;
    float acc = 0.f;
#pragma unroll
    for (int it = 0; it < 4; ++it) {
        int e = it * 128 + lane * 4;
        float4 r  = *(const float4*)(row + e);
        float4 hh = *(const float4*)(hv + e);
        acc += r.x * hh.x + r.y * hh.y + r.z * hh.z + r.w * hh.w;
    }
    for (int off = 16; off > 0; off >>= 1) acc += __shfl_down(acc, off, 32);
    if (lane == 0) {
        scores[(size_t)b * SS + s] = mask[(size_t)b * SS + s]
                                         ? -__builtin_inff() : acc;
    }
}

// ---------------------------------------------------------------------------
// K5: softmax over S per batch row (in-place)
// ---------------------------------------------------------------------------
__global__ void softmax_kernel(float* __restrict__ scores) {
    __shared__ float red[256];
    int b = blockIdx.x, t = threadIdx.x;
    float* row = scores + (size_t)b * SS;
    float mx = -__builtin_inff();
    for (int i = t; i < SS; i += 256) mx = fmaxf(mx, row[i]);
    red[t] = mx; __syncthreads();
    for (int o = 128; o > 0; o >>= 1) { if (t < o) red[t] = fmaxf(red[t], red[t + o]); __syncthreads(); }
    mx = red[0]; __syncthreads();
    float sum = 0.f;
    for (int i = t; i < SS; i += 256) { float e = __expf(row[i] - mx); row[i] = e; sum += e; }
    red[t] = sum; __syncthreads();
    for (int o = 128; o > 0; o >>= 1) { if (t < o) red[t] += red[t + o]; __syncthreads(); }
    float inv = 1.f / red[0];
    for (int i = t; i < SS; i += 256) row[i] *= inv;
}

// ---------------------------------------------------------------------------
// K6: ctx[b,e] += sum_{s in chunk} w[b,s] * src[b,s,e]   (atomic partial sums)
// ---------------------------------------------------------------------------
__global__ void ctx_kernel(const float* __restrict__ src,
                           const float* __restrict__ attw,
                           float* __restrict__ ctx) {
    int e = threadIdx.x;            // 512
    int b = blockIdx.x;             // 32
    int chunk = blockIdx.y;         // 16 chunks of 128 rows
    const float* w = attw + (size_t)b * SS + chunk * 128;
    const float* base = src + ((size_t)b * SS + chunk * 128) * EE + e;
    float acc = 0.f;
    for (int s = 0; s < 128; ++s) acc += w[s] * base[(size_t)s * EE];
    atomicAdd(&ctx[(size_t)b * EE + e], acc);
}

// ---------------------------------------------------------------------------
extern "C" void kernel_launch(void* const* d_in, const int* in_sizes, int n_in,
                              void* d_out, int out_size, void* d_ws, size_t ws_size,
                              hipStream_t stream) {
    const float* enc_src     = (const float*)d_in[0];
    const float* enc_src_att = (const float*)d_in[1];
    const float* enc_col     = (const float*)d_in[2];
    const float* enc_tab     = (const float*)d_in[3];
    const unsigned char* src_mask = (const unsigned char*)d_in[4]; // bool8
    const float* att   = (const float*)d_in[5];
    const float* h0    = (const float*)d_in[6];
    const float* c0    = (const float*)d_in[7];
    const int* head_agg  = (const int*)d_in[8];
    const int* head_col  = (const int*)d_in[9];
    const int* quant_tab = (const int*)d_in[10];
    const int* box_op    = (const int*)d_in[11];
    const float* agg_emb    = (const float*)d_in[12];
    const float* box_op_emb = (const float*)d_in[13];
    const float* W_box = (const float*)d_in[14];
    const float* b_box = (const float*)d_in[15];
    const float* W_ih  = (const float*)d_in[16];
    const float* W_hh  = (const float*)d_in[17];
    const float* b_ih  = (const float*)d_in[18];
    const float* b_hh  = (const float*)d_in[19];
    const float* W_att = (const float*)d_in[20];
    const float* b_att = (const float*)d_in[21];

    float* ws = (float*)d_ws;
    float* head_emb = ws;                 // [32,512]
    float* body_emb = ws + 16384;         // [32,512]
    float* boxop_e  = ws + 32768;         // [32,512]
    float* enc_box  = ws + 49152;         // [32,512]
    float* gates    = ws + 65536;         // [32,2048]
    float* hbuf     = ws + 131072;        // [32,512]
    float* scores   = ws + 147456;        // [32,2048]
    float* ctx      = ws + 212992;        // [32,512]

    // K0: gathers + means
    embed_kernel<<<BB, EE, 0, stream>>>(enc_col, enc_tab, head_agg, head_col,
                                        quant_tab, box_op, agg_emb, box_op_emb,
                                        head_emb, body_emb, boxop_e);

    // K1: encoded_box = relu([head|body|boxop] @ W_box.T + b_box)
    // tiles = 2 * (512/16) = 64 -> 16 blocks x 4 waves
    wmma_gemm_kernel<<<16, 128, 0, stream>>>(head_emb, body_emb, boxop_e,
                                             W_box, 3 * EE, /*kSplit=*/3 * EE,
                                             W_box, 3 * EE,
                                             b_box, nullptr,
                                             /*Ktot=*/3 * EE, /*N=*/EE,
                                             enc_box, /*relu=*/1);

    // K2: gates = [enc_box|att|h0] @ [W_ih|W_hh].T   (biases deferred to K3)
    // tiles = 2 * (2048/16) = 256 -> 64 blocks x 4 waves
    wmma_gemm_kernel<<<64, 128, 0, stream>>>(enc_box, att, h0,
                                             W_ih, 2 * EE, /*kSplit=*/2 * EE,
                                             W_hh, EE,
                                             nullptr, nullptr,
                                             /*Ktot=*/3 * EE, /*N=*/4 * EE,
                                             gates, /*relu=*/0);

    // K3: LSTM cell
    lstm_cell_kernel<<<(BB * EE) / 256, 256, 0, stream>>>(gates, b_ih, b_hh, c0, hbuf);

    // K4: attention scores (dominant stream: 134 MB)
    scores_kernel<<<dim3(SS / 8, BB), 256, 0, stream>>>(enc_src_att, hbuf, src_mask, scores);

    // K5: masked softmax
    softmax_kernel<<<BB, 256, 0, stream>>>(scores);

    // K6: context (second 134 MB stream), atomic partials into zeroed ctx
    hipMemsetAsync(ctx, 0, (size_t)BB * EE * sizeof(float), stream);
    ctx_kernel<<<dim3(BB, 16), EE, 0, stream>>>(enc_src, scores, ctx);

    // K7: att_new = relu([h|ctx] @ W_att.T + b_att) -> d_out
    wmma_gemm_kernel<<<16, 128, 0, stream>>>(hbuf, ctx, hbuf,
                                             W_att, 2 * EE, /*kSplit=*/2 * EE,
                                             W_att, 2 * EE,
                                             b_att, nullptr,
                                             /*Ktot=*/2 * EE, /*N=*/EE,
                                             (float*)d_out, /*relu=*/1);
}
